// HEAD_30348238914073
// MI455X (gfx1250) — compile-verified, matrix operand order checked by reading, not run on hardware
//
#include <hip/hip_runtime.h>

typedef __attribute__((ext_vector_type(16))) __bf16        v16bf;
typedef __attribute__((ext_vector_type(8)))  float         v8f;
typedef __attribute__((ext_vector_type(4)))  unsigned int  u32x4;
typedef __attribute__((ext_vector_type(4)))  float         f32x4;

union Frag { u32x4 u[2]; v16bf v; };

__device__ __forceinline__ unsigned short f32_to_bf16(float f) {
  unsigned int u = __float_as_uint(f);
  unsigned int r = u + 0x7FFFu + ((u >> 16) & 1u);   // round-to-nearest-even
  return (unsigned short)(r >> 16);
}

__device__ __forceinline__ v8f wmma_bf16(const Frag& A, const Frag& B, v8f C) {
  return __builtin_amdgcn_wmma_f32_16x16x32_bf16(false, A.v, false, B.v,
                                                 (short)0, C, false, false);
}

// Issue both halves of one 32-byte B fragment as ds_load_b128 with immediate
// DS offsets (16-bit field, max used = 64528).  asm volatile => never hoisted
// out of the persistent loop, never turned into flat/scratch traffic.
template <int OFF>
__device__ __forceinline__ void lds_frag_load(unsigned base, Frag& B) {
  asm volatile("ds_load_b128 %0, %2 offset:%c3\n\t"
               "ds_load_b128 %1, %2 offset:%c4"
               : "=v"(B.u[0]), "=v"(B.u[1])
               : "v"(base), "n"(OFF), "n"(OFF + 16));
}

// Wait until only the most recently issued fragment (2 DS ops) is in flight;
// "+v" ties make the consuming WMMAs depend on this wait.
__device__ __forceinline__ void lds_wait_keep2(Frag& B) {
  asm volatile("s_wait_dscnt 0x2" : "+v"(B.u[0]), "+v"(B.u[1]));
}
__device__ __forceinline__ void lds_wait_all(Frag& B) {
  asm volatile("s_wait_dscnt 0x0" : "+v"(B.u[0]), "+v"(B.u[1]));
}

// nt-loop (template-unrolled), software-pipelined: load frag nt+1, wait for
// frag nt, fire two WMMAs (tile0 + tile1) per fragment.
template <int KS, int NT>
__device__ __forceinline__ void nt_steps(unsigned base, const Frag& A0, const Frag& A1,
                                         v8f* acc, Frag& Bcur) {
  if constexpr (NT < 7) {
    Frag Bnext;
    lds_frag_load<((NT + 1) * 8 + KS) * 1024>(base, Bnext);
    lds_wait_keep2(Bcur);
    acc[NT]     = wmma_bf16(A0, Bcur, acc[NT]);
    acc[8 + NT] = wmma_bf16(A1, Bcur, acc[8 + NT]);
    nt_steps<KS, NT + 1>(base, A0, A1, acc, Bnext);
  } else {
    lds_wait_all(Bcur);
    acc[NT]     = wmma_bf16(A0, Bcur, acc[NT]);
    acc[8 + NT] = wmma_bf16(A1, Bcur, acc[8 + NT]);
  }
}

// ks-loop (template-unrolled): gather the two A fragments for this k-step
// (K = ks*32.. of the 256-wide [row||col] concat) then run the 8 n-tiles.
template <int KS>
__device__ __forceinline__ void ks_steps(unsigned base,
                                         const unsigned short* __restrict__ fr0,
                                         const unsigned short* __restrict__ fc0,
                                         const unsigned short* __restrict__ fr1,
                                         const unsigned short* __restrict__ fc1,
                                         int kk, v8f* acc) {
  const unsigned short* a0p = (KS < 4 ? fr0 : fc0) + (KS & 3) * 32 + kk;
  const unsigned short* a1p = (KS < 4 ? fr1 : fc1) + (KS & 3) * 32 + kk;
  Frag A0, A1;
  A0.u[0] = *(const u32x4*)(a0p);
  A0.u[1] = *(const u32x4*)(a0p + 16);
  A1.u[0] = *(const u32x4*)(a1p);
  A1.u[1] = *(const u32x4*)(a1p + 16);
  Frag B0;
  lds_frag_load<KS * 1024>(base, B0);      // fragment (nt=0, KS)
  nt_steps<KS, 0>(base, A0, A1, acc, B0);
  if constexpr (KS < 7) ks_steps<KS + 1>(base, fr0, fc0, fr1, fc1, kk, acc);
}

// ---------------------------------------------------------------------------
// Pre-pass 1: swizzle W1 [256,128] f32 -> bf16 WMMA B-fragments.
// Fragment f = nt*8+ks is a 32x16 bf16 tile (K=32, N=16): per lane 16 bf16,
// mirroring the A layout: lanes 0-15 hold K={0..7,16..23}, lanes 16-31 hold
// K={8..15,24..31}, N = nt*16 + lane%16.  Also zero the per-block partials.
// ---------------------------------------------------------------------------
__global__ __launch_bounds__(256) void prep_w1(const float* __restrict__ W1,
                                               unsigned short* __restrict__ frag,
                                               float* __restrict__ partials,
                                               int nPartials) {
  int t = blockIdx.x * 256 + threadIdx.x;
  if (t < 64 * 512) {
    int fragid = t >> 9;                 // 0..63
    int lane   = (t >> 4) & 31;
    int elem   = t & 15;
    int nt = fragid >> 3, ks = fragid & 7;
    int N  = nt * 16 + (lane & 15);
    int K  = ks * 32 + ((lane < 16) ? 0 : 8) + ((elem < 8) ? elem : (8 + elem));
    frag[t] = f32_to_bf16(W1[K * 128 + N]);
  }
  if (t < nPartials) partials[t] = 0.0f;
}

// ---------------------------------------------------------------------------
// Pre-pass 2: feature table f32 -> bf16 (25.6 MB, stays L2-resident).
// ---------------------------------------------------------------------------
__global__ __launch_bounds__(256) void cvt_feature(const float* __restrict__ f,
                                                   unsigned short* __restrict__ o,
                                                   int n8) {
  int i = blockIdx.x * 256 + threadIdx.x;
  if (i >= n8) return;
  const f32x4* src = (const f32x4*)f + (size_t)i * 2;
  f32x4 a = src[0], b = src[1];
  union { unsigned short s[8]; u32x4 u; } out;
  out.s[0] = f32_to_bf16(a[0]); out.s[1] = f32_to_bf16(a[1]);
  out.s[2] = f32_to_bf16(a[2]); out.s[3] = f32_to_bf16(a[3]);
  out.s[4] = f32_to_bf16(b[0]); out.s[5] = f32_to_bf16(b[1]);
  out.s[6] = f32_to_bf16(b[2]); out.s[7] = f32_to_bf16(b[3]);
  *((u32x4*)o + i) = out.u;
}

// ---------------------------------------------------------------------------
// Per-16-edge-tile epilogue: bias + PReLU + x W2[128,2] head + masked CE.
// C layout: N = lane%16, VGPR j = row j (lanes 0-15) / row j+8 (lanes 16-31).
// Row j is owned by lane j (its M=j); row j+8 by lane 24+j (its M=j+8).
// ---------------------------------------------------------------------------
__device__ __forceinline__ float tile_epilogue(const v8f* acc, int lane, int M,
                                               const float* __restrict__ b1,
                                               const float* __restrict__ alpha,
                                               const float* __restrict__ W2,
                                               float b20, float b21,
                                               int r, int c,
                                               const int* __restrict__ label,
                                               bool ve) {
  float p0[8], p1[8];
#pragma unroll
  for (int j = 0; j < 8; ++j) { p0[j] = 0.0f; p1[j] = 0.0f; }
#pragma unroll
  for (int nt = 0; nt < 8; ++nt) {
    const int ncol = nt * 16 + M;
    const float bias = b1[ncol], al = alpha[ncol];
    const float w0 = W2[2 * ncol], w1 = W2[2 * ncol + 1];
#pragma unroll
    for (int j = 0; j < 8; ++j) {
      float h = acc[nt][j] + bias;
      h = (h >= 0.0f) ? h : al * h;
      p0[j] += h * w0;
      p1[j] += h * w1;
    }
  }
#pragma unroll
  for (int mask = 1; mask < 16; mask <<= 1) {
#pragma unroll
    for (int j = 0; j < 8; ++j) {
      p0[j] += __shfl_xor(p0[j], mask);
      p1[j] += __shfl_xor(p1[j], mask);
    }
  }
  const int lr = label[r], lc = label[c];
  float ll = 0.0f;
#pragma unroll
  for (int j = 0; j < 8; ++j) {
    const bool act = (lane == j) || (lane == 24 + j);
    float l0 = p0[j] + b20, l1 = p1[j] + b21;
    float mx  = fmaxf(l0, l1);
    float lse = mx + __logf(__expf(l0 - mx) + __expf(l1 - mx));
    float ly  = (lr == lc) ? l1 : l0;
    ll += (act && ve) ? (lse - ly) : 0.0f;
  }
  return ll;
}

// ---------------------------------------------------------------------------
// Main kernel: one wave = one 32-edge group (two 16-row M-tiles); each W1
// B-fragment ds_load feeds TWO v_wmma_f32_16x16x32_bf16 ops; 128 WMMAs per
// group iteration, LDS reads software-pipelined against the XDL pipe.
// ---------------------------------------------------------------------------
__global__ __launch_bounds__(256) void edge_mlp_wmma(
    const unsigned short* __restrict__ featb,
    const unsigned short* __restrict__ w1frag,
    const float* __restrict__ b1, const float* __restrict__ alpha,
    const float* __restrict__ W2, const float* __restrict__ b2,
    const int* __restrict__ row, const int* __restrict__ col,
    const int* __restrict__ label,
    float* __restrict__ partials, int nGroups, int nEdges) {
  __shared__ unsigned short ldsB[64 * 512];   // 64 KB of W1 fragments
  __shared__ float wsum[8];

  const int lane = threadIdx.x & 31;
  const int wave = threadIdx.x >> 5;
  const int M    = lane & 15;                 // row-within-tile this lane serves

  // Stage W1 fragments into LDS once per (persistent) block.
  {
    const u32x4* src = (const u32x4*)w1frag;
    u32x4*       dst = (u32x4*)ldsB;
    for (int i = threadIdx.x; i < (64 * 512) / 8; i += 256) dst[i] = src[i];
  }
  __syncthreads();

  // 32-bit DS address of this lane's slice of fragment 0 (generic LDS
  // pointers carry the DS offset in their low 32 bits).
  const unsigned ldsBase = (unsigned)(size_t)(void*)ldsB + (unsigned)lane * 32u;

  const float b20 = b2[0], b21 = b2[1];
  const int   kk  = (lane < 16) ? 0 : 8;      // A/B lane-half K offset
  float lloss = 0.0f;

  const int groupStride = gridDim.x * 8;
  for (int g = blockIdx.x * 8 + wave; g < nGroups; g += groupStride) {
    const long eBase = (long)g * 32;
    const long e0 = eBase + M;                // tile 0 row M
    const long e1 = eBase + 16 + M;           // tile 1 row M
    const bool ve0 = (e0 < nEdges), ve1 = (e1 < nEdges);
    const long ec0 = ve0 ? e0 : (long)(nEdges - 1);
    const long ec1 = ve1 ? e1 : (long)(nEdges - 1);
    const int r0 = row[ec0], c0 = col[ec0];
    const int r1 = row[ec1], c1 = col[ec1];
    const unsigned short* fr0 = featb + (size_t)r0 * 128;
    const unsigned short* fc0 = featb + (size_t)c0 * 128;
    const unsigned short* fr1 = featb + (size_t)r1 * 128;
    const unsigned short* fc1 = featb + (size_t)c1 * 128;

    v8f acc[16];                              // [0..7]=tile0, [8..15]=tile1
#pragma unroll
    for (int i = 0; i < 16; ++i) acc[i] = (v8f){};

    ks_steps<0>(ldsBase, fr0, fc0, fr1, fc1, kk, acc);

    lloss += tile_epilogue(acc,     lane, M, b1, alpha, W2, b20, b21, r0, c0, label, ve0);
    lloss += tile_epilogue(acc + 8, lane, M, b1, alpha, W2, b20, b21, r1, c1, label, ve1);
  }

  // Deterministic block reduction: wave butterfly -> fixed-order LDS sum.
#pragma unroll
  for (int mask = 16; mask >= 1; mask >>= 1) lloss += __shfl_xor(lloss, mask);
  if (lane == 0) wsum[wave] = lloss;
  __syncthreads();
  if (threadIdx.x == 0) {
    float s = 0.0f;
#pragma unroll
    for (int w = 0; w < 8; ++w) s += wsum[w];
    partials[blockIdx.x] = s;
  }
}

// ---------------------------------------------------------------------------
// Final deterministic reduction of per-block partials -> mean loss.
// ---------------------------------------------------------------------------
__global__ __launch_bounds__(256) void finalize_loss(const float* __restrict__ partials,
                                                     int n, float invE,
                                                     float* __restrict__ out) {
  __shared__ float s[256];
  float acc = 0.0f;
  for (int i = threadIdx.x; i < n; i += 256) acc += partials[i];
  s[threadIdx.x] = acc;
  __syncthreads();
  for (int st = 128; st > 0; st >>= 1) {
    if ((int)threadIdx.x < st) s[threadIdx.x] += s[threadIdx.x + st];
    __syncthreads();
  }
  if (threadIdx.x == 0) out[0] = s[0] * invE;
}

extern "C" void kernel_launch(void* const* d_in, const int* in_sizes, int n_in,
                              void* d_out, int out_size, void* d_ws, size_t ws_size,
                              hipStream_t stream) {
  const float* feature = (const float*)d_in[0];
  const float* W1      = (const float*)d_in[1];
  const float* b1      = (const float*)d_in[2];
  const float* alpha   = (const float*)d_in[3];
  const float* W2      = (const float*)d_in[4];
  const float* b2      = (const float*)d_in[5];
  const int*   row     = (const int*)d_in[6];
  const int*   col     = (const int*)d_in[7];
  const int*   label   = (const int*)d_in[8];

  const int nEdges  = in_sizes[6];
  const int nNodes  = in_sizes[0] / 128;
  const int nGroups = (nEdges + 31) / 32;
  const int nBlocks = 1024;

  // Workspace layout: [W1 frags bf16: 64KB][feature bf16: nNodes*256B][partials]
  unsigned short* w1frag = (unsigned short*)d_ws;
  unsigned short* featb  = (unsigned short*)((char*)d_ws + 65536);
  float* partials = (float*)((char*)d_ws + 65536 + (size_t)nNodes * 128 * 2);

  prep_w1<<<(64 * 512 + 255) / 256, 256, 0, stream>>>(W1, w1frag, partials, nBlocks);
  const int n8 = nNodes * 16;  // (nNodes*128)/8 vectors of 8 floats
  cvt_feature<<<(n8 + 255) / 256, 256, 0, stream>>>(feature, featb, n8);
  edge_mlp_wmma<<<nBlocks, 256, 0, stream>>>(featb, w1frag, b1, alpha, W2, b2,
                                             row, col, label, partials, nGroups, nEdges);
  finalize_loss<<<1, 256, 0, stream>>>(partials, nBlocks, 1.0f / (float)nEdges,
                                       (float*)d_out);
}
